// ConvSelfAttentionModule_84731114815740
// MI455X (gfx1250) — compile-verified, matrix operand order checked by reading, not run on hardware
//
#include <hip/hip_runtime.h>

#define BB   4
#define CCH  256
#define CQKD 128
#define NNP  4096
#define NSEG 4
#define LOG2E 1.4426950408889634f

typedef __attribute__((ext_vector_type(16))) __bf16 v16bf;
typedef __attribute__((ext_vector_type(8)))  float  v8f;

union Frag16 { uint4 q[2]; v16bf v; };

__device__ __forceinline__ float fast_exp2(float x) {
    return __builtin_amdgcn_exp2f(x);   // v_exp_f32 (base-2 native)
}

// Load a 16-element bf16 fragment: elements [0..7] from row[0..7], [8..15] from row[16..23].
// Matches the CDNA5 16-bit A/B fragment K-pattern when 'row' is offset by half*8.
__device__ __forceinline__ v16bf load_frag_b16(const __bf16* __restrict__ row) {
    Frag16 f;
    const uint4* p = reinterpret_cast<const uint4*>(row);
    f.q[0] = p[0];   // K +0..7   (8 bf16 = 16B)
    f.q[1] = p[2];   // K +16..23 (8 bf16 = 16B)
    return f.v;
}

__device__ __forceinline__ v8f wmma_bf16(v16bf a, v16bf b, v8f c) {
    return __builtin_amdgcn_wmma_f32_16x16x32_bf16(false, a, false, b, (short)0, c, false, false);
}

// ---------------------------------------------------------------------------
// Kernel 1: projection GEMM (templated on #o-tiles and output layout).
// TRANSPOSED=1: out[b][n][o] (bf16, for Q^T/K^T); TRANSPOSED=0: out[b][o][n] (V).
// One wave per (b, 16-wide n-tile); wave computes all NT o-tiles.
// ---------------------------------------------------------------------------
template <int NT, int TRANSPOSED>
__global__ __launch_bounds__(128) void proj_kernel(
    const float* __restrict__ x, const float* __restrict__ W,
    const float* __restrict__ bias, __bf16* __restrict__ outp)
{
    const int lane = threadIdx.x & 31;
    const int wid  = blockIdx.x * (blockDim.x >> 5) + (threadIdx.x >> 5);
    const int b    = wid >> 8;
    const int n0   = (wid & 255) << 4;
    const int hf   = lane >> 4;
    const int ln   = lane & 15;

    v8f acc[NT];
    #pragma unroll
    for (int t = 0; t < NT; ++t) acc[t] = (v8f)0.f;

    const float* xb = x + (size_t)b * CCH * NNP;

    for (int kc = 0; kc < 8; ++kc) {
        const int d0 = kc * 32 + hf * 8;
        // B fragment from x [C][N] (strided gather, f32 -> bf16 on the fly)
        v16bf bfr;
        #pragma unroll
        for (int e = 0; e < 8; ++e)
            bfr[e] = (__bf16)xb[(size_t)(d0 + e) * NNP + n0 + ln];
        #pragma unroll
        for (int e = 0; e < 8; ++e)
            bfr[8 + e] = (__bf16)xb[(size_t)(d0 + 16 + e) * NNP + n0 + ln];

        #pragma unroll
        for (int ot = 0; ot < NT; ++ot) {
            const float* wrow = W + (ot * 16 + ln) * CCH + d0;
            v16bf afr;
            #pragma unroll
            for (int e = 0; e < 8; ++e) afr[e]     = (__bf16)wrow[e];
            #pragma unroll
            for (int e = 0; e < 8; ++e) afr[8 + e] = (__bf16)wrow[16 + e];
            acc[ot] = wmma_bf16(afr, bfr, acc[ot]);
        }
    }

    #pragma unroll
    for (int ot = 0; ot < NT; ++ot) {
        #pragma unroll
        for (int r = 0; r < 8; ++r) {
            const int o = ot * 16 + hf * 8 + r;   // D layout: m = r + hf*8
            float v = acc[ot][r] + bias[o];
            if (TRANSPOSED)
                outp[((size_t)b * NNP + n0 + ln) * CQKD + o] = (__bf16)v;
            else
                outp[((size_t)b * (NT * 16) + o) * NNP + n0 + ln] = (__bf16)v;
        }
    }
}

// ---------------------------------------------------------------------------
// Kernel 2: row stats in log2 domain. t(i,j) = s(i,j)*log2(e).
// m_i = max_j t, Z_i = sum_j 2^(t-m). One wave per (b, 16-row i-tile);
// per-lane online pass over its j subset, 4-step xor-shuffle merge per half.
// ---------------------------------------------------------------------------
__global__ __launch_bounds__(128) void rowstats_kernel(
    const __bf16* __restrict__ QT, const __bf16* __restrict__ KT,
    float* __restrict__ mrow, float* __restrict__ rzrow)
{
    const int lane = threadIdx.x & 31;
    const int wid  = blockIdx.x * (blockDim.x >> 5) + (threadIdx.x >> 5);
    const int b    = wid >> 8;
    const int i0   = (wid & 255) << 4;
    const int hf   = lane >> 4;
    const int ln   = lane & 15;

    // A fragments (rows i0..i0+15 of Q^T), resident for the whole j sweep
    v16bf aq[4];
    #pragma unroll
    for (int kc = 0; kc < 4; ++kc)
        aq[kc] = load_frag_b16(QT + ((size_t)b * NNP + i0 + ln) * CQKD + kc * 32 + hf * 8);

    float mrun[8], srun[8];
    #pragma unroll
    for (int r = 0; r < 8; ++r) { mrun[r] = -1e30f; srun[r] = 0.f; }

    const __bf16* KTb = KT + (size_t)b * NNP * CQKD;
    for (int jt = 0; jt < NNP / 16; ++jt) {
        v8f s = (v8f)0.f;
        #pragma unroll
        for (int kc = 0; kc < 4; ++kc) {
            v16bf bk = load_frag_b16(KTb + (size_t)(jt * 16 + ln) * CQKD + kc * 32 + hf * 8);
            s = wmma_bf16(aq[kc], bk, s);
        }
        #pragma unroll
        for (int r = 0; r < 8; ++r) {
            float t  = s[r] * LOG2E;
            float nm = fmaxf(mrun[r], t);
            srun[r]  = srun[r] * fast_exp2(mrun[r] - nm) + fast_exp2(t - nm);
            mrun[r]  = nm;
        }
    }

    // merge across the 16 lanes of each half (lane^mask stays within the half)
    #pragma unroll
    for (int mask = 1; mask <= 8; mask <<= 1) {
        #pragma unroll
        for (int r = 0; r < 8; ++r) {
            float om = __shfl_xor(mrun[r], mask, 32);
            float os = __shfl_xor(srun[r], mask, 32);
            float nm = fmaxf(mrun[r], om);
            srun[r]  = srun[r] * fast_exp2(mrun[r] - nm) + os * fast_exp2(om - nm);
            mrun[r]  = nm;
        }
    }
    if (ln == 0) {
        #pragma unroll
        for (int r = 0; r < 8; ++r) {
            const int i = i0 + hf * 8 + r;     // D layout row id
            mrow[b * NNP + i]  = mrun[r];
            rzrow[b * NNP + i] = 1.0f / srun[r];
        }
    }
}

// ---------------------------------------------------------------------------
// Kernel 3: af[:,j] += V[:,i-range] * P[i-range,j], P = 2^(t-m)*rz.
// Wave decode: the 4 waves of a block share (b, i-segment) and differ in jt,
// so QT fragments, all 16 V fragments and the stats vectors are identical
// addresses across the block -> WGP-cache hits; only KT (loop-invariant,
// register-resident) differs per wave.
// ---------------------------------------------------------------------------
__global__ __launch_bounds__(128) void attnv_kernel(
    const __bf16* __restrict__ QT, const __bf16* __restrict__ KT,
    const __bf16* __restrict__ Vb,
    const float* __restrict__ mrow, const float* __restrict__ rzrow,
    float* __restrict__ af)
{
    const int lane = threadIdx.x & 31;
    const int wid  = blockIdx.x * (blockDim.x >> 5) + (threadIdx.x >> 5);
    const int b    = wid / (NSEG * 256);
    const int rem  = wid % (NSEG * 256);
    const int seg  = rem / 256;          // shared within a block
    const int jt   = rem % 256;          // varies across the block's 4 waves
    const int j0   = jt * 16;
    const int hf   = lane >> 4;
    const int ln   = lane & 15;
    const int ibeg = seg * (NNP / NSEG);
    const int iend = ibeg + (NNP / NSEG);

    // K^T B-fragments for this j-tile: resident across the whole i loop
    v16bf bk[4];
    #pragma unroll
    for (int kc = 0; kc < 4; ++kc)
        bk[kc] = load_frag_b16(KT + ((size_t)b * NNP + j0 + ln) * CQKD + kc * 32 + hf * 8);

    v8f acc[16];
    #pragma unroll
    for (int t = 0; t < 16; ++t) acc[t] = (v8f)0.f;

    const __bf16* QTb = QT + (size_t)b * NNP * CQKD;
    const __bf16* Vbb = Vb + (size_t)b * CCH * NNP;
    const float*  mb  = mrow  + b * NNP;
    const float*  zb  = rzrow + b * NNP;

    for (int ib = ibeg; ib < iend; ib += 32) {
        v8f s0 = (v8f)0.f, s1 = (v8f)0.f;
        #pragma unroll
        for (int kc = 0; kc < 4; ++kc) {
            v16bf a0 = load_frag_b16(QTb + (size_t)(ib + ln) * CQKD + kc * 32 + hf * 8);
            s0 = wmma_bf16(a0, bk[kc], s0);
            v16bf a1 = load_frag_b16(QTb + (size_t)(ib + 16 + ln) * CQKD + kc * 32 + hf * 8);
            s1 = wmma_bf16(a1, bk[kc], s1);
        }

        // stats for the 8+8 rows this lane-half owns: contiguous -> float4 loads
        const int ibase = ib + hf * 8;
        float m0[8], z0[8], m1[8], z1[8];
        *(float4*)(m0)     = *(const float4*)(mb + ibase);
        *(float4*)(m0 + 4) = *(const float4*)(mb + ibase + 4);
        *(float4*)(m1)     = *(const float4*)(mb + ibase + 16);
        *(float4*)(m1 + 4) = *(const float4*)(mb + ibase + 20);
        *(float4*)(z0)     = *(const float4*)(zb + ibase);
        *(float4*)(z0 + 4) = *(const float4*)(zb + ibase + 4);
        *(float4*)(z1)     = *(const float4*)(zb + ibase + 16);
        *(float4*)(z1 + 4) = *(const float4*)(zb + ibase + 20);

        // P tile -> B fragment (K = i, N = j): lane already holds column j.
        // elements 0..7  <- s0 rows (i-local r + hf*8)  == K {0..7}/{8..15}
        // elements 8..15 <- s1 rows (i-local 16 + ...)  == K {16..23}/{24..31}
        v16bf pb;
        #pragma unroll
        for (int r = 0; r < 8; ++r) {
            float p0 = fast_exp2(__builtin_fmaf(s0[r], LOG2E, -m0[r])) * z0[r];
            float p1 = fast_exp2(__builtin_fmaf(s1[r], LOG2E, -m1[r])) * z1[r];
            pb[r]     = (__bf16)p0;
            pb[8 + r] = (__bf16)p1;
        }

        #pragma unroll
        for (int ct = 0; ct < 16; ++ct) {
            v16bf va = load_frag_b16(Vbb + (size_t)(ct * 16 + ln) * NNP + ib + hf * 8);
            acc[ct] = wmma_bf16(va, pb, acc[ct]);
        }
    }

    #pragma unroll
    for (int ct = 0; ct < 16; ++ct) {
        #pragma unroll
        for (int r = 0; r < 8; ++r) {
            const int c = ct * 16 + hf * 8 + r;
            unsafeAtomicAdd(&af[((size_t)b * CCH + c) * NNP + j0 + ln], acc[ct][r]);
        }
    }
}

// ---------------------------------------------------------------------------
// Kernel 4: out = gamma * af + x  (float4 vectorized)
// ---------------------------------------------------------------------------
__global__ __launch_bounds__(256) void epilogue_kernel(
    const float* __restrict__ af, const float* __restrict__ x,
    const float* __restrict__ gamma, float* __restrict__ out)
{
    const size_t idx = (size_t)blockIdx.x * blockDim.x + threadIdx.x;
    const float g = gamma[0];
    const float4 a  = ((const float4*)af)[idx];
    const float4 xx = ((const float4*)x)[idx];
    float4 o;
    o.x = g * a.x + xx.x;  o.y = g * a.y + xx.y;
    o.z = g * a.z + xx.z;  o.w = g * a.w + xx.w;
    ((float4*)out)[idx] = o;
}

extern "C" void kernel_launch(void* const* d_in, const int* in_sizes, int n_in,
                              void* d_out, int out_size, void* d_ws, size_t ws_size,
                              hipStream_t stream) {
    const float* x     = (const float*)d_in[0];
    const float* wq    = (const float*)d_in[1];
    const float* bq    = (const float*)d_in[2];
    const float* wk    = (const float*)d_in[3];
    const float* bk    = (const float*)d_in[4];
    const float* wv    = (const float*)d_in[5];
    const float* bv    = (const float*)d_in[6];
    const float* gamma = (const float*)d_in[7];
    float* out = (float*)d_out;

    char* ws = (char*)d_ws;
    size_t off = 0;
    auto alloc = [&](size_t bytes) -> void* {
        void* p = ws + off;
        off = (off + bytes + 255) & ~(size_t)255;
        return p;
    };
    __bf16* QT    = (__bf16*)alloc((size_t)BB * NNP * CQKD * 2);
    __bf16* KT    = (__bf16*)alloc((size_t)BB * NNP * CQKD * 2);
    __bf16* Vb    = (__bf16*)alloc((size_t)BB * CCH * NNP * 2);
    float*  mrow  = (float*)alloc((size_t)BB * NNP * 4);
    float*  rzrow = (float*)alloc((size_t)BB * NNP * 4);
    float*  af    = (float*)alloc((size_t)BB * CCH * NNP * 4);

    hipMemsetAsync(af, 0, (size_t)BB * CCH * NNP * sizeof(float), stream);

    // projections: 1024 waves each (B * N/16), 4 waves/block
    proj_kernel<8, 1><<<256, 128, 0, stream>>>(x, wq, bq, QT);
    proj_kernel<8, 1><<<256, 128, 0, stream>>>(x, wk, bk, KT);
    proj_kernel<16, 0><<<256, 128, 0, stream>>>(x, wv, bv, Vb);
    // 1024 waves, 4/block
    rowstats_kernel<<<256, 128, 0, stream>>>(QT, KT, mrow, rzrow);
    // BB*NSEG*256 = 4096 waves, 4/block
    attnv_kernel<<<1024, 128, 0, stream>>>(QT, KT, Vb, mrow, rzrow, af);
    // BB*CCH*NNP / 4 float4s
    epilogue_kernel<<<(BB * CCH * NNP / 4) / 256, 256, 0, stream>>>(af, x, gamma, out);
}
